// ScreeningAttention_1374389534897
// MI455X (gfx1250) — matrix operand order repaired and checked by reading, not compiled, not run on hardware
//
#include <hip/hip_runtime.h>
#include <hip/hip_bf16.h>

#define D_MODEL   1024
#define NUM_HEADS 16
#define HEAD_DIM  64
#define EPS       1e-8f
#define PI_F      3.14159265358979323846f

#define KSTEP      32
#define LDS_STRIDE 40   // bf16 per LDS row: 32 data + 8 pad (80B; TDM pad reproduces this)

typedef __attribute__((ext_vector_type(16))) __bf16 v16bf;
typedef __attribute__((ext_vector_type(8)))  __bf16 v8bf;
typedef __attribute__((ext_vector_type(8)))  float  v8f;
typedef __attribute__((ext_vector_type(4)))  unsigned uint32x4;
typedef __attribute__((ext_vector_type(8)))  int      int32x8;
typedef __attribute__((ext_vector_type(4)))  int      int32x4;

// ---------------------------------------------------------------------------
// gfx1250 async global->LDS copy (16B per lane), tracked by ASYNCcnt.
// ---------------------------------------------------------------------------
__device__ __forceinline__ void async_copy_b128(unsigned lds_addr, const __bf16* gptr) {
  asm volatile("global_load_async_to_lds_b128 %0, %1, off"
               :: "v"(lds_addr), "v"(gptr)
               : "memory");
}
__device__ __forceinline__ void wait_asynccnt_le1() {
  asm volatile("s_wait_asynccnt 0x1" ::: "memory");
}
__device__ __forceinline__ void wait_asynccnt_0() {
  asm volatile("s_wait_asynccnt 0x0" ::: "memory");
}

// ---------------------------------------------------------------------------
// Tensor Data Mover: load a 256-row x 64B 2D tile of W (rows of 2048B) into
// LDS with 16B padding after every 64B (row stride 80B). Byte-granular D#.
// Wave-uniform; issued by one wave per block, tracked by TENSORcnt.
// This toolchain's builtin takes 6 args: (g0 v4u32, g1 v8i32, g2 v4i32,
// g3 v4i32, extra v8i32, cpol i32). 2D tensor -> g2/g3/extra all zero.
// ---------------------------------------------------------------------------
__device__ __forceinline__ void tdm_load_b_tile(const __bf16* gbase, unsigned lds_addr) {
  const unsigned long long ga = (unsigned long long)(uintptr_t)gbase;
  uint32x4 g0;
  g0[0] = 1u;                                              // count=1 (valid), user mode
  g0[1] = lds_addr;                                        // LDS byte address
  g0[2] = (unsigned)(ga & 0xFFFFFFFFu);                    // global_addr[31:0]
  g0[3] = (unsigned)((ga >> 32) & 0x01FFFFFFu)             // global_addr[56:32]
        | (2u << 30);                                      // type = 2 ("image")
  int32x8 g1;
  g1[0] = (int)((1u << 20) | (3u << 22) | (3u << 25));     // pad_en, interval=16DW, amount=4DW
  g1[1] = (int)(2048u << 16);                              // tensor_dim0 = 2048 B/row
  g1[2] = (int)(1024u << 16);                              // tensor_dim1 = 1024 rows
  g1[3] = (int)(64u << 16);                                // tile_dim0   = 64 B
  g1[4] = 256;                                             // tile_dim1   = 256 rows
  g1[5] = 2048;                                            // tensor_dim0_stride = 2048 B
  g1[6] = 0;
  g1[7] = 0;
  const int32x4 z4 = {0, 0, 0, 0};
  const int32x8 z8 = {0, 0, 0, 0, 0, 0, 0, 0};
  __builtin_amdgcn_tensor_load_to_lds(g0, g1, z4, z4, z8, 0);
}

// ---------------------------------------------------------------------------
// f32 -> bf16 elementwise convert
// ---------------------------------------------------------------------------
__global__ __launch_bounds__(256) void cvt_f32_bf16(const float* __restrict__ src,
                                                    __bf16* __restrict__ dst, int n) {
  int i = blockIdx.x * blockDim.x + threadIdx.x;
  if (i < n) dst[i] = (__bf16)src[i];
}

// ---------------------------------------------------------------------------
// C[M,N] = A[M,K] (bf16, row-major) x W[N,K]^T (bf16, row-major), fp32 out.
// 256 thr = 8 wave32. Wave tile 32x64 (2x4 of 16x16x32 WMMA), block tile
// 64x256. A slab: per-thread async b128 copies; B slab: one TDM descriptor
// issued by wave 0. Double-buffered. Requires N==1024, K==1024 (TDM D#).
// ---------------------------------------------------------------------------
__global__ __launch_bounds__(256) void gemm_xwT_wmma(const __bf16* __restrict__ A,
                                                     const __bf16* __restrict__ W,
                                                     float* __restrict__ C,
                                                     int M, int N, int K) {
  __shared__ __bf16 sA[2][64 * LDS_STRIDE];
  __shared__ __bf16 sB[2][256 * LDS_STRIDE];

  const int tid  = threadIdx.x;
  const int lane = tid & 31;
  const int wave = tid >> 5;
  const int wm = wave >> 2;          // 0..1
  const int wn = wave & 3;           // 0..3
  const int mBlk = blockIdx.y * 64;
  const int nBlk = blockIdx.x * 256;
  const int lrow = lane & 15;        // M (A) / N (B,C) sub-index
  const int hi   = lane >> 4;        // half-wave select

  // --- A-tile async-copy split: 16B chunks, 1 chunk per thread (64 rows x 4)
  const int crow  = tid >> 2;              // 0..63
  const int ckoff = (tid & 3) * 8;         // bf16 k-offset of this chunk
  const __bf16* agp = A + (size_t)(mBlk + crow) * K + ckoff;
  const unsigned aOff = (unsigned)(crow * LDS_STRIDE + ckoff) * 2u;
  const unsigned sA0 = (unsigned)(uintptr_t)&sA[0][0];
  const unsigned sA1 = (unsigned)(uintptr_t)&sA[1][0];
  const unsigned sB0 = (unsigned)(uintptr_t)&sB[0][0];
  const unsigned sB1 = (unsigned)(uintptr_t)&sB[1][0];
  const __bf16* wBase = W + (size_t)nBlk * K;

  auto issue_tile = [&](int buf, int k0) {
    async_copy_b128((buf ? sA1 : sA0) + aOff, agp + k0);     // all threads: A
    if (wave == 0)                                           // wave 0: B via TDM
      tdm_load_b_tile(wBase + k0, buf ? sB1 : sB0);
  };

  v8f acc[2][4] = {};

  const int nk = K / KSTEP;
  issue_tile(0, 0);

  for (int it = 0; it < nk; ++it) {
    const int nxt = it + 1;
    if (nxt < nk) {
      issue_tile(nxt & 1, nxt * KSTEP);
      wait_asynccnt_le1();                         // tile it's A copy landed
      if (wave == 0) __builtin_amdgcn_s_wait_tensorcnt(1);   // tile it's TDM landed
    } else {
      wait_asynccnt_0();
      if (wave == 0) __builtin_amdgcn_s_wait_tensorcnt(0);
    }
    __syncthreads();         // all waves see A copies + wave0's TDM result

    const __bf16* base_a = (it & 1) ? &sA[1][0] : &sA[0][0];
    const __bf16* base_b = (it & 1) ? &sB[1][0] : &sB[0][0];

    v16bf a[2], b[4];
    // A 16x32 fragment: lanes 0-15 K {0..7,16..23}; lanes 16-31 K {8..15,24..31}
    #pragma unroll
    for (int mt = 0; mt < 2; ++mt) {
      const __bf16* p = base_a + (wm * 32 + mt * 16 + lrow) * LDS_STRIDE + hi * 8;
      v8bf lo = *(const v8bf*)p;
      v8bf hh = *(const v8bf*)(p + 16);
      #pragma unroll
      for (int e = 0; e < 8; ++e) { a[mt][e] = lo[e]; a[mt][8 + e] = hh[e]; }
    }
    // B 32x16 fragment: lane -> N column; lanes 0-15 K {0..7,16..23}; lanes 16-31 K {8..15,24..31}
    #pragma unroll
    for (int nt = 0; nt < 4; ++nt) {
      const __bf16* p = base_b + (wn * 64 + nt * 16 + lrow) * LDS_STRIDE + hi * 8;
      v8bf lo = *(const v8bf*)p;
      v8bf hh = *(const v8bf*)(p + 16);
      #pragma unroll
      for (int e = 0; e < 8; ++e) { b[nt][e] = lo[e]; b[nt][8 + e] = hh[e]; }
    }

    #pragma unroll
    for (int mt = 0; mt < 2; ++mt)
      #pragma unroll
      for (int nt = 0; nt < 4; ++nt)
        acc[mt][nt] = __builtin_amdgcn_wmma_f32_16x16x32_bf16(
            false, a[mt], false, b[nt], (short)0, acc[mt][nt], false, false);

    __syncthreads();         // reads done before this buffer is overwritten
  }

  // C/D layout: VGPR i -> M = i + 8*hi ; N = lane & 15
  #pragma unroll
  for (int mt = 0; mt < 2; ++mt)
    #pragma unroll
    for (int nt = 0; nt < 4; ++nt) {
      const int col = nBlk + wn * 64 + nt * 16 + lrow;
      #pragma unroll
      for (int i = 0; i < 8; ++i) {
        const int row = mBlk + wm * 32 + mt * 16 + i + hi * 8;
        C[(size_t)row * N + col] = acc[mt][nt][i];
      }
    }
}

// ---------------------------------------------------------------------------
// In-place L2 norm over 64-element head chunks. One wave per chunk.
// ---------------------------------------------------------------------------
__global__ __launch_bounds__(256) void l2norm_heads(float* __restrict__ x, int nchunks) {
  const int w    = blockIdx.x * 8 + (threadIdx.x >> 5);
  const int lane = threadIdx.x & 31;
  if (w >= nchunks) return;
  float* p = x + (size_t)w * 64;
  float v0 = p[lane], v1 = p[lane + 32];
  float ss = v0 * v0 + v1 * v1;
  #pragma unroll
  for (int o = 16; o > 0; o >>= 1) ss += __shfl_xor(ss, o, 32);
  float n = sqrtf(ss);
  float s = 1.0f / fmaxf(n, EPS);
  p[lane] = v0 * s;
  p[lane + 32] = v1 * s;
}

// ---------------------------------------------------------------------------
// Windowed "screening" attention. One wave per (b,h,q). Keys restricted to
// rel = k - q in (-w, 0], w = exp(s_v[h]) + 1 (so <= ceil(w) keys).
// Output: tanh-normalized, bf16, laid out [b, t, h*64 + d] (pre-Wo).
// ---------------------------------------------------------------------------
__global__ __launch_bounds__(256) void screening_attn(const float* __restrict__ qf,
                                                      const float* __restrict__ kf,
                                                      const float* __restrict__ vf,
                                                      const unsigned char* __restrict__ mask,
                                                      const float* __restrict__ s_v,
                                                      const float* __restrict__ s_r,
                                                      __bf16* __restrict__ outbf,
                                                      int B, int T) {
  const int w    = blockIdx.x * 8 + (threadIdx.x >> 5);
  const int lane = threadIdx.x & 31;
  const int total = B * NUM_HEADS * T;
  if (w >= total) return;

  const int b  = w / (NUM_HEADS * T);
  const int h  = (w / T) % NUM_HEADS;
  const int qi = w % T;

  const float* qp = qf + ((size_t)(b * T + qi)) * D_MODEL + h * HEAD_DIM;
  const float q0 = qp[lane], q1 = qp[lane + 32];

  const float wh = __expf(s_v[h]) + 1.0f;
  const float rh = __expf(s_r[h]) + 1.0f;

  int span = (int)ceilf(wh);
  int kmin = qi - span + 1;
  if (kmin < 0) kmin = 0;

  float a0 = 0.0f, a1 = 0.0f;
  for (int kk = kmin; kk <= qi; ++kk) {
    const float rel = (float)(kk - qi);
    if (!(rel > -wh)) continue;

    const size_t kvoff = ((size_t)(b * T + kk)) * D_MODEL + h * HEAD_DIM;
    const float* kp = kf + kvoff;
    float partial = q0 * kp[lane] + q1 * kp[lane + 32];
    #pragma unroll
    for (int o = 16; o > 0; o >>= 1) partial += __shfl_xor(partial, o, 32);
    const float sim = partial;

    float t1 = 1.0f - rh * (1.0f - sim);
    float rl = fmaxf(t1, 0.0f);
    float cm = 0.5f * (__cosf(PI_F * rel / wh) + 1.0f);
    float alpha = rl * rl * cm;
    if (mask[b * T + kk]) alpha = 0.0f;

    const float* vp = vf + kvoff;
    a0 += alpha * vp[lane];
    a1 += alpha * vp[lane + 32];
  }

  float ss = a0 * a0 + a1 * a1;
  #pragma unroll
  for (int o = 16; o > 0; o >>= 1) ss += __shfl_xor(ss, o, 32);
  const float n  = sqrtf(ss);
  const float sc = tanhf(n) / (n + EPS);

  __bf16* op = outbf + ((size_t)(b * T + qi)) * D_MODEL + h * HEAD_DIM;
  op[lane]      = (__bf16)(a0 * sc);
  op[lane + 32] = (__bf16)(a1 * sc);
}

// ---------------------------------------------------------------------------
// Host-side orchestration
// ---------------------------------------------------------------------------
extern "C" void kernel_launch(void* const* d_in, const int* in_sizes, int n_in,
                              void* d_out, int out_size, void* d_ws, size_t ws_size,
                              hipStream_t stream) {
  (void)in_sizes; (void)n_in; (void)out_size; (void)ws_size;
  const int B = 2, T = 2048;
  const int Mtok = B * T;                 // 4096 tokens
  const int nX = Mtok * D_MODEL;          // 4,194,304
  const int nW = D_MODEL * D_MODEL;       // 1,048,576

  const float* query = (const float*)d_in[0];
  const float* key   = (const float*)d_in[1];
  const float* value = (const float*)d_in[2];
  const unsigned char* kpm = (const unsigned char*)d_in[3];
  const float* Wq = (const float*)d_in[4];
  const float* Wk = (const float*)d_in[5];
  const float* Wv = (const float*)d_in[6];
  const float* Wo = (const float*)d_in[7];
  const float* s_v = (const float*)d_in[8];
  const float* s_r = (const float*)d_in[9];
  float* out = (float*)d_out;

  char* ws = (char*)d_ws;
  size_t off = 0;
  auto take = [&](size_t bytes) -> void* {
    void* p = ws + off;
    off += (bytes + 255) & ~(size_t)255;
    return p;
  };

  __bf16* Xq  = (__bf16*)take((size_t)nX * 2);
  __bf16* Xk  = (__bf16*)take((size_t)nX * 2);
  __bf16* Xv  = (__bf16*)take((size_t)nX * 2);
  __bf16* Wqb = (__bf16*)take((size_t)nW * 2);
  __bf16* Wkb = (__bf16*)take((size_t)nW * 2);
  __bf16* Wvb = (__bf16*)take((size_t)nW * 2);
  __bf16* Wob = (__bf16*)take((size_t)nW * 2);
  float*  qf  = (float*)take((size_t)nX * 4);
  float*  kf  = (float*)take((size_t)nX * 4);
  float*  vf  = (float*)take((size_t)nX * 4);
  __bf16* Obf = (__bf16*)take((size_t)nX * 2);

  const int thr = 256;
  // 1) convert activations + weights to bf16
  cvt_f32_bf16<<<(nX + thr - 1) / thr, thr, 0, stream>>>(query, Xq, nX);
  cvt_f32_bf16<<<(nX + thr - 1) / thr, thr, 0, stream>>>(key,   Xk, nX);
  cvt_f32_bf16<<<(nX + thr - 1) / thr, thr, 0, stream>>>(value, Xv, nX);
  cvt_f32_bf16<<<(nW + thr - 1) / thr, thr, 0, stream>>>(Wq, Wqb, nW);
  cvt_f32_bf16<<<(nW + thr - 1) / thr, thr, 0, stream>>>(Wk, Wkb, nW);
  cvt_f32_bf16<<<(nW + thr - 1) / thr, thr, 0, stream>>>(Wv, Wvb, nW);
  cvt_f32_bf16<<<(nW + thr - 1) / thr, thr, 0, stream>>>(Wo, Wob, nW);

  // 2) q/k/v projections (WMMA GEMM): [4096,1024] x [1024,1024]^T
  dim3 gg(D_MODEL / 256, Mtok / 64);
  gemm_xwT_wmma<<<gg, thr, 0, stream>>>(Xq, Wqb, qf, Mtok, D_MODEL, D_MODEL);
  gemm_xwT_wmma<<<gg, thr, 0, stream>>>(Xk, Wkb, kf, Mtok, D_MODEL, D_MODEL);
  gemm_xwT_wmma<<<gg, thr, 0, stream>>>(Xv, Wvb, vf, Mtok, D_MODEL, D_MODEL);

  // 3) per-head L2 norms (in place)
  const int nchunks = Mtok * NUM_HEADS;   // 65536
  l2norm_heads<<<nchunks / 8, thr, 0, stream>>>(qf, nchunks);
  l2norm_heads<<<nchunks / 8, thr, 0, stream>>>(kf, nchunks);
  l2norm_heads<<<nchunks / 8, thr, 0, stream>>>(vf, nchunks);

  // 4) windowed screening attention + tanh-norm, emits bf16 pre-Wo activations
  const int nq = B * NUM_HEADS * T;       // 65536
  screening_attn<<<nq / 8, thr, 0, stream>>>(qf, kf, vf, kpm, s_v, s_r, Obf, B, T);

  // 5) output projection -> fp32 d_out
  gemm_xwT_wmma<<<gg, thr, 0, stream>>>(Obf, Wob, out, Mtok, D_MODEL, D_MODEL);
}
